// MoEMLP_257698038435
// MI455X (gfx1250) — compile-verified, matrix operand order checked by reading, not run on hardware
//
#include <hip/hip_runtime.h>

// ---------------------------------------------------------------------------
// Fused MoE MLP (dense-all-experts with top-2 routed weighting), bf16 WMMA.
//   x:            [4096, 1024] f32
//   gate_w:       [8, 1024] f32
//   gate_up_proj: [8, 1024, 2048] f32   (row = H index = K of GEMM1)
//   down_proj:    [8, 1024, 1024] f32   (row = D index = K of GEMM2)
//   out:          [4096, 1024] f32
//
// Pass 1: prepack weights fp32 -> bf16 WMMA B-fragments into d_ws (~50 MB).
// Pass 2: fused router + 8-expert SwiGLU MLP. One workgroup (512 thr = 16
//         wave32) per 32-token tile: two A-fragments per K-step so every
//         B-fragment load feeds 2 WMMAs (halves L2 B-traffic per token).
//         Routing weight is folded into the GEMM2 A-matrix so GEMM2
//         accumulates directly into the persistent output accumulators.
// ---------------------------------------------------------------------------

namespace {

constexpr int kH    = 1024;
constexpr int kD    = 1024;
constexpr int k2D   = 2048;
constexpr int kE    = 8;
constexpr int kMT   = 32;     // token tile (2 row-fragments of 16)
constexpr int kNTok = 4096;   // B*T

// fragment geometry: one B-fragment = 32x16 tile = 32 lanes x 16 bf16 = 1 KB
constexpr int kKB    = kH / 32;          // 32 K-blocks (both GEMMs, K=1024)
constexpr int kNTGU  = k2D / 16;         // 128 column tiles (gate_up)
constexpr int kNTDN  = kH / 16;          // 64 column tiles (down)
constexpr size_t kFragShorts = 32 * 16;  // shorts per fragment
constexpr size_t kFragsGU = (size_t)kE * kKB * kNTGU;   // 32768
constexpr size_t kFragsDN = (size_t)kE * kKB * kNTDN;   // 16384
constexpr size_t kPackGUShorts = kFragsGU * kFragShorts;
constexpr size_t kPackDNShorts = kFragsDN * kFragShorts;
constexpr size_t kPackBytes = (kPackGUShorts + kPackDNShorts) * sizeof(unsigned short);

typedef __attribute__((ext_vector_type(16))) __bf16 v16bf;
typedef __attribute__((ext_vector_type(8)))  float  v8f;

__device__ __forceinline__ unsigned short f32_to_bf16(float f) {
  union { float f; unsigned int u; } a; a.f = f;
  const unsigned int r = a.u + 0x7FFFu + ((a.u >> 16) & 1u);   // RNE
  return (unsigned short)(r >> 16);
}

__device__ __forceinline__ v8f wmma_bf16(v16bf a, v16bf b, v8f c) {
  return __builtin_amdgcn_wmma_f32_16x16x32_bf16(
      /*neg_a=*/false, a, /*neg_b=*/false, b,
      /*c_mod=*/(short)0, c, /*reuse_a=*/false, /*reuse_b=*/false);
}

// K<->element mapping mirrors the 16-bit A-matrix layout (ISA 7.12.2):
// lanes 0-15 hold K {0..7,16..23}, lanes 16-31 hold K {8..15,24..31}.
__device__ __forceinline__ int k_of_elem(int ee, int half) {
  return (ee & 7) + (half << 3) + ((ee >= 8) ? 16 : 0);
}

// Slow path: build a B-fragment from fp32 weights in registers.
__device__ __forceinline__ v16bf load_b_tile_f32(const float* __restrict__ Wk,
                                                 int ldc, int col, int half) {
  union { v16bf v; unsigned short u[16]; } b;
#pragma unroll
  for (int ee = 0; ee < 16; ++ee)
    b.u[ee] = f32_to_bf16(Wk[(size_t)k_of_elem(ee, half) * ldc + col]);
  return b.v;
}

} // namespace

// ---------------------------------------------------------------------------
// Pass 1: pre-pack fp32 weights into bf16 WMMA B-fragments in workspace.
// ---------------------------------------------------------------------------
__global__ __launch_bounds__(256) void moe_prepack_kernel(
    const float* __restrict__ gup, const float* __restrict__ gdn,
    unsigned short* __restrict__ pgu, unsigned short* __restrict__ pdn) {
  const size_t gtid = (size_t)blockIdx.x * blockDim.x + threadIdx.x;
  const size_t frag = gtid >> 5;
  if (frag >= kFragsGU + kFragsDN) return;
  const int lane = (int)(gtid & 31);
  const int half = lane >> 4;
  const int nc   = lane & 15;

  const float* W;
  unsigned short* dst;
  int ldn, ntile, kb;
  if (frag < kFragsGU) {
    const size_t fr = frag;
    const int e = (int)(fr / ((size_t)kKB * kNTGU));
    const int r = (int)(fr % ((size_t)kKB * kNTGU));
    kb = r >> 7; ntile = r & (kNTGU - 1);
    W = gup + (size_t)e * kH * k2D;  ldn = k2D;
    dst = pgu + fr * kFragShorts;
  } else {
    const size_t fr = frag - kFragsGU;
    const int e = (int)(fr / ((size_t)kKB * kNTDN));
    const int r = (int)(fr % ((size_t)kKB * kNTDN));
    kb = r >> 6; ntile = r & (kNTDN - 1);
    W = gdn + (size_t)e * kD * kH;   ldn = kH;
    dst = pdn + fr * kFragShorts;
  }
  const int col = ntile * 16 + nc;
  const int row0 = kb * 32;
  union { v16bf v; unsigned short u[16]; } b;
#pragma unroll
  for (int ee = 0; ee < 16; ++ee)
    b.u[ee] = f32_to_bf16(W[(size_t)(row0 + k_of_elem(ee, half)) * ldn + col]);
  *(v16bf*)(dst + (size_t)lane * 16) = b.v;   // coalesced 32 B per lane
}

// ---------------------------------------------------------------------------
// Pass 2: fused router + 8-expert SwiGLU MLP with routed-weight accumulation.
// 512 threads = 16 wave32 per workgroup; 32-token tile (2 A row-fragments).
// ---------------------------------------------------------------------------
template <bool PACKED>
__global__ __launch_bounds__(512) void moe_mlp_fused_kernel(
    const float* __restrict__ xg,
    const float* __restrict__ gate_w,
    const float* __restrict__ gup,
    const float* __restrict__ gdn,
    const unsigned short* __restrict__ pgu,
    const unsigned short* __restrict__ pdn,
    float* __restrict__ outg) {
  // A-layout packed tiles per row-fragment: [rf][k-block][lane][16] = 64 KB ea
  __shared__ __align__(32) unsigned short xpack[2][32][32][16];
  __shared__ __align__(32) unsigned short hpack[2][32][32][16];
  __shared__ float rw_lds[kMT][kE];
  __shared__ float logit_lds[kMT][kE];

  const int tid  = threadIdx.x;
  const int wv_  = tid >> 5;    // wave id 0..15
  const int lane = tid & 31;
  const int nl   = lane & 15;   // column within 16-wide tile
  const int half = lane >> 4;   // lane group (selects row/K half)
  const int tok0 = blockIdx.x * kMT;

  // ---- phase 0: pack x tile fp32 -> bf16 into WMMA A layout --------------
  {
    const int m  = tid >> 4;            // token row 0..31
    const int rf = m >> 4;              // row fragment
    const int mm = m & 15;
    const int ks = tid & 15;            // 64-wide K chunk
    const float* xr = xg + (size_t)(tok0 + m) * kH + ks * 64;
#pragma unroll 2
    for (int kk = 0; kk < 64; kk += 4) {
      const float4 v4 = *(const float4*)(xr + kk);
      const float vals[4] = {v4.x, v4.y, v4.z, v4.w};
#pragma unroll
      for (int q = 0; q < 4; ++q) {
        const int k    = ks * 64 + kk + q;
        const int kb   = k >> 5;
        const int ko   = k & 31;
        const int lidx = mm + (((ko >> 3) & 1) << 4);
        const int ep   = (ko & 7) + ((ko >= 16) ? 8 : 0);
        xpack[rf][kb][lidx][ep] = f32_to_bf16(vals[q]);
      }
    }
  }

  // ---- phase R: router logits, softmax, top-2, double-eps renorm ---------
  if (tid < kMT * kE) {
    const int m = tid >> 3;
    const int e = tid & 7;
    const float* xr = xg + (size_t)(tok0 + m) * kH;
    const float* gr = gate_w + (size_t)e * kH;
    float acc = 0.f;
    for (int k = 0; k < kH; k += 4)
      acc += xr[k] * gr[k] + xr[k + 1] * gr[k + 1] +
             xr[k + 2] * gr[k + 2] + xr[k + 3] * gr[k + 3];
    logit_lds[m][e] = acc;
  }
  __syncthreads();
  if (tid < kMT) {
    const int m = tid;
    float p[kE];
    float mx = logit_lds[m][0];
#pragma unroll
    for (int e = 1; e < kE; ++e) mx = fmaxf(mx, logit_lds[m][e]);
    float s = 0.f;
#pragma unroll
    for (int e = 0; e < kE; ++e) { p[e] = __expf(logit_lds[m][e] - mx); s += p[e]; }
    const float inv = 1.f / s;
#pragma unroll
    for (int e = 0; e < kE; ++e) p[e] *= inv;
    int i1 = 0; float w1 = p[0];
#pragma unroll
    for (int e = 1; e < kE; ++e) if (p[e] > w1) { w1 = p[e]; i1 = e; }
    int i2 = (i1 == 0) ? 1 : 0; float w2 = p[i2];
#pragma unroll
    for (int e = 0; e < kE; ++e) if (e != i1 && p[e] > w2) { w2 = p[e]; i2 = e; }
    const float t = w1 + w2 + 1e-9f;
    w1 /= t; w2 /= t;
    const float t2 = w1 + w2 + 1e-9f;
#pragma unroll
    for (int e = 0; e < kE; ++e) rw_lds[m][e] = 0.f;
    rw_lds[m][i1] = w1 / t2;
    rw_lds[m][i2] = w2 / t2;
  }
  __syncthreads();

  // Persistent output accumulators: wave owns 4 column tiles x 2 row frags
  v8f oacc[4][2];
#pragma unroll
  for (int j = 0; j < 4; ++j) { v8f z = {}; oacc[j][0] = z; oacc[j][1] = z; }

  for (int e = 0; e < kE; ++e) {
    const float* Wg = gup + (size_t)e * kH * k2D;   // slow path
    const float* Wd = gdn + (size_t)e * kD * kH;    // slow path
    const v16bf* Pg = (const v16bf*)pgu + (size_t)e * kKB * kNTGU * 32;
    const v16bf* Pd = (const v16bf*)pdn + (size_t)e * kKB * kNTDN * 32;

    // hpack is re-used per expert: wait for previous expert's GEMM2 readers
    __syncthreads();

    // routing weights for the rows this lane's C-fragments cover
    float wrow[2][8];
#pragma unroll
    for (int r = 0; r < 2; ++r)
#pragma unroll
      for (int v = 0; v < 8; ++v)
        wrow[r][v] = rw_lds[r * 16 + half * 8 + v][e];

    // ---- phase 1+2: gate_up GEMM (K=1024), SiLU(gate)*up*rw -> hpack -----
#pragma unroll 1
    for (int cc = 0; cc < 2; ++cc) {
      const int gt0 = wv_ * 4 + cc * 2;              // first gate tile
      v8f ga[2][2], ua[2][2];                        // [colTile][rowFrag]
#pragma unroll
      for (int j = 0; j < 2; ++j) {
        v8f z = {};
        ga[j][0] = z; ga[j][1] = z; ua[j][0] = z; ua[j][1] = z;
      }
#pragma unroll 1
      for (int kb = 0; kb < 32; ++kb) {
        const v16bf a0 = *(const v16bf*)&xpack[0][kb][lane][0];
        const v16bf a1 = *(const v16bf*)&xpack[1][kb][lane][0];
        v16bf bg[2], bu[2];
        if (PACKED) {
          const v16bf* Pk = Pg + ((size_t)kb * kNTGU) * 32 + lane;
          __builtin_prefetch(Pk + (size_t)kNTGU * 32, 0, 1);  // next K-block
#pragma unroll
          for (int j = 0; j < 2; ++j) {
            bg[j] = Pk[(size_t)(gt0 + j) * 32];        // gate tiles
            bu[j] = Pk[(size_t)(64 + gt0 + j) * 32];   // up tiles
          }
        } else {
          const float* Wk = Wg + (size_t)(kb * 32) * k2D;
#pragma unroll
          for (int j = 0; j < 2; ++j) {
            bg[j] = load_b_tile_f32(Wk, k2D, (gt0 + j) * 16 + nl, half);
            bu[j] = load_b_tile_f32(Wk, k2D, kD + (gt0 + j) * 16 + nl, half);
          }
        }
        // each B-fragment feeds two WMMAs (row fragments) from registers
#pragma unroll
        for (int j = 0; j < 2; ++j) {
          ga[j][0] = wmma_bf16(a0, bg[j], ga[j][0]);
          ga[j][1] = wmma_bf16(a1, bg[j], ga[j][1]);
          ua[j][0] = wmma_bf16(a0, bu[j], ua[j][0]);
          ua[j][1] = wmma_bf16(a1, bu[j], ua[j][1]);
        }
      }
      // SiLU(gate)*up scaled by routing weight, repack as GEMM2 A-matrix
#pragma unroll
      for (int j = 0; j < 2; ++j) {
        const int q    = (gt0 + j) * 16 + nl;          // 0..1023
        const int kb2  = q >> 5;
        const int ko   = q & 31;
        const int loff = ((ko >> 3) & 1) << 4;
        const int ep   = (ko & 7) + ((ko >= 16) ? 8 : 0);
#pragma unroll
        for (int r = 0; r < 2; ++r) {
          const int lidx = half * 8 + loff;            // + v below
#pragma unroll
          for (int v = 0; v < 8; ++v) {
            const float g = ga[j][r][v];
            const float u = ua[j][r][v];
            const float sig = __builtin_amdgcn_rcpf(1.f + __expf(-g));
            const float hval = g * sig * u * wrow[r][v];
            hpack[r][kb2][lidx + v][ep] = f32_to_bf16(hval);
          }
        }
      }
    }
    __syncthreads();

    // ---- phase 3: down GEMM (K=1024) accumulating straight into oacc -----
    {
      const int nt0 = wv_ * 4;
#pragma unroll 1
      for (int kb = 0; kb < 32; ++kb) {
        const v16bf a0 = *(const v16bf*)&hpack[0][kb][lane][0];
        const v16bf a1 = *(const v16bf*)&hpack[1][kb][lane][0];
        v16bf bfrag[4];
        if (PACKED) {
          const v16bf* Pk = Pd + ((size_t)kb * kNTDN) * 32 + lane;
          __builtin_prefetch(Pk + (size_t)kNTDN * 32, 0, 1);
#pragma unroll
          for (int j = 0; j < 4; ++j)
            bfrag[j] = Pk[(size_t)(nt0 + j) * 32];
        } else {
          const float* Wk = Wd + (size_t)(kb * 32) * kH;
#pragma unroll
          for (int j = 0; j < 4; ++j)
            bfrag[j] = load_b_tile_f32(Wk, kH, (nt0 + j) * 16 + nl, half);
        }
#pragma unroll
        for (int j = 0; j < 4; ++j) {
          oacc[j][0] = wmma_bf16(a0, bfrag[j], oacc[j][0]);
          oacc[j][1] = wmma_bf16(a1, bfrag[j], oacc[j][1]);
        }
      }
    }
  }

  // ---- phase 4: store out tile (each value written exactly once) ---------
#pragma unroll
  for (int j = 0; j < 4; ++j) {
    const int cb = (wv_ * 4 + j) * 16 + nl;
#pragma unroll
    for (int r = 0; r < 2; ++r) {
#pragma unroll
      for (int v = 0; v < 8; ++v) {
        const int m = r * 16 + half * 8 + v;
        outg[(size_t)(tok0 + m) * kH + cb] = oacc[j][r][v];
      }
    }
  }
}

extern "C" void kernel_launch(void* const* d_in, const int* in_sizes, int n_in,
                              void* d_out, int out_size, void* d_ws, size_t ws_size,
                              hipStream_t stream) {
  (void)in_sizes; (void)n_in; (void)out_size;
  const float* x   = (const float*)d_in[0];   // [4096,1024]
  const float* gw  = (const float*)d_in[1];   // [8,1024]
  const float* gup = (const float*)d_in[2];   // [8,1024,2048]
  const float* gdn = (const float*)d_in[3];   // [8,1024,1024]
  float* out = (float*)d_out;                 // [4096,1024]

  dim3 grid(kNTok / kMT);  // 128 workgroups, one per 32-token tile
  dim3 block(512);         // 16 wave32

  if (ws_size >= kPackBytes) {
    // Fast path: bf16-prepacked weight fragments in workspace.
    unsigned short* pgu = (unsigned short*)d_ws;
    unsigned short* pdn = pgu + kPackGUShorts;
    const size_t nThreads = (kFragsGU + kFragsDN) * 32;
    const int nBlocks = (int)((nThreads + 255) / 256);
    moe_prepack_kernel<<<nBlocks, 256, 0, stream>>>(gup, gdn, pgu, pdn);
    moe_mlp_fused_kernel<true><<<grid, block, 0, stream>>>(
        x, gw, gup, gdn, pgu, pdn, out);
  } else {
    // Fallback: convert B-tiles in registers (no workspace needed).
    moe_mlp_fused_kernel<false><<<grid, block, 0, stream>>>(
        x, gw, gup, gdn, nullptr, nullptr, out);
  }
}
